// EncoderRNN_84482006713136
// MI455X (gfx1250) — matrix operand order repaired and checked by reading, not compiled, not run on hardware
//
#include <hip/hip_runtime.h>
#include <hip/hip_bf16.h>
#include <stdint.h>

#define H    1024
#define E    1024
#define S    4096
#define G4H  4096            // 4*H gate rows
#define NBLK_REC 128         // persistent blocks in recurrent phase
#define UPB  8               // hidden units per recurrent block (H / NBLK_REC)
#define WSTRIDE 1032         // padded LDS row stride (floats): 1032 % 64 = 8 -> no bank conflicts

typedef __attribute__((ext_vector_type(16))) __bf16         v16bf;
typedef __attribute__((ext_vector_type(8)))  float          v8f;
typedef __attribute__((ext_vector_type(8)))  unsigned short u16x8;

__device__ __forceinline__ unsigned short f2bf(float f) {
  unsigned u = __float_as_uint(f);
  u += 0x7fffu + ((u >> 16) & 1u);          // round-to-nearest-even
  return (unsigned short)(u >> 16);
}

__device__ __forceinline__ float sigmoidf(float x) { return 1.f / (1.f + __expf(-x)); }

// ---------------------------------------------------------------- init
__global__ void init_state(float* hb0, float* hb1, unsigned* sync) {
  int i = blockIdx.x * blockDim.x + threadIdx.x;
  if (i < H) { hb0[i] = 0.f; hb1[i] = 0.f; }
  if (i < 2) sync[i] = 0u;
}

// ------------------------------------------- embedding gather + ReLU -> bf16
__global__ void embed_relu_bf16(const int* __restrict__ tok,
                                const float* __restrict__ emb,
                                unsigned short* __restrict__ Xb) {
  int t = blockIdx.x;
  long row = (long)tok[t];
  const float* src = emb + row * E;
  unsigned short* dst = Xb + (long)t * E;
  for (int e = threadIdx.x; e < E; e += blockDim.x) {
    float v = src[e];
    dst[e] = f2bf(v > 0.f ? v : 0.f);
  }
}

// ---------------------------------------------------------------- f32 -> bf16
__global__ void cvt_bf16(const float* __restrict__ src,
                         unsigned short* __restrict__ dst, int n) {
  int i = blockIdx.x * blockDim.x + threadIdx.x;
  if (i < n) dst[i] = f2bf(src[i]);
}

// ---------------- G_ih = X(4096x1024) * W_ih^T  (bf16 WMMA, 16x64 per wave)
// Each wave owns a 16x64 output strip: one A fragment feeds 4 accumulators,
// cutting global traffic from 64 B/WMMA to 40 B/WMMA and raising WMMA density.
__global__ void __launch_bounds__(256)
gemm_ih_wmma(const unsigned short* __restrict__ Xb,   // S x E, row major
             const unsigned short* __restrict__ Wb,   // 4H x E, row major
             float* __restrict__ Gih) {               // S x 4H, row major
  const int lane = threadIdx.x & 31;
  const int wave = threadIdx.x >> 5;
  const int task = blockIdx.x * 8 + wave;             // 256 M-tiles x 64 N-strips
  const int tM = (task >> 6) << 4;
  const int tN = (task & 63) << 6;                    // 64-wide N strip
  const int hi  = lane >> 4;                          // half-wave select
  const int l15 = lane & 15;

  // A (16x32 bf16): lane<16 rows M=l15 hold K {k..k+15};
  //                 lane>=16 same rows hold K {k+16..k+31}.
  const unsigned short* arow  = Xb + (long)(tM + l15) * E + (hi ? 16 : 0);
  // B (32x16 bf16): lane holds 16 contiguous K at column N; lanes>=16 offset +16.
  const unsigned short* brow0 = Wb + (long)(tN +  0 + l15) * E + (hi ? 16 : 0);
  const unsigned short* brow1 = Wb + (long)(tN + 16 + l15) * E + (hi ? 16 : 0);
  const unsigned short* brow2 = Wb + (long)(tN + 32 + l15) * E + (hi ? 16 : 0);
  const unsigned short* brow3 = Wb + (long)(tN + 48 + l15) * E + (hi ? 16 : 0);

  v8f acc0 = {}, acc1 = {}, acc2 = {}, acc3 = {};
  for (int k = 0; k < E; k += 32) {
    union { u16x8 h[2]; v16bf v; } A, B0, B1, B2, B3;
    A.h[0]  = *(const u16x8*)(arow  + k);
    A.h[1]  = *(const u16x8*)(arow  + k + 8);
    B0.h[0] = *(const u16x8*)(brow0 + k);
    B0.h[1] = *(const u16x8*)(brow0 + k + 8);
    B1.h[0] = *(const u16x8*)(brow1 + k);
    B1.h[1] = *(const u16x8*)(brow1 + k + 8);
    B2.h[0] = *(const u16x8*)(brow2 + k);
    B2.h[1] = *(const u16x8*)(brow2 + k + 8);
    B3.h[0] = *(const u16x8*)(brow3 + k);
    B3.h[1] = *(const u16x8*)(brow3 + k + 8);
    if (k + 32 < E) {
      __builtin_prefetch(arow  + k + 32, 0, 3);       // near-cache prefetch
      __builtin_prefetch(brow0 + k + 32, 0, 3);
      __builtin_prefetch(brow2 + k + 32, 0, 3);
    }
    acc0 = __builtin_amdgcn_wmma_f32_16x16x32_bf16(false, A.v, false, B0.v,
                                                   (short)0, acc0, false, false);
    acc1 = __builtin_amdgcn_wmma_f32_16x16x32_bf16(false, A.v, false, B1.v,
                                                   (short)0, acc1, false, false);
    acc2 = __builtin_amdgcn_wmma_f32_16x16x32_bf16(false, A.v, false, B2.v,
                                                   (short)0, acc2, false, false);
    acc3 = __builtin_amdgcn_wmma_f32_16x16x32_bf16(false, A.v, false, B3.v,
                                                   (short)0, acc3, false, false);
  }

  // C/D layout: VGPR r -> row (tM + r + 8*hi), lane l15 -> col
  float* drow = Gih + (long)(tM + (hi ? 8 : 0)) * G4H + tN + l15;
#pragma unroll
  for (int r = 0; r < 8; ++r) {
    drow[(long)r * G4H +  0] = acc0[r];
    drow[(long)r * G4H + 16] = acc1[r];
    drow[(long)r * G4H + 32] = acc2[r];
    drow[(long)r * G4H + 48] = acc3[r];
  }
}

// ---------------------------------------------------------------- grid barrier
__device__ __forceinline__ void grid_sync(unsigned* cnt, unsigned* gen, unsigned nblk) {
  __syncthreads();
  if (threadIdx.x == 0) {
    __threadfence();
    unsigned g = __hip_atomic_load(gen, __ATOMIC_RELAXED, __HIP_MEMORY_SCOPE_AGENT);
    unsigned arrived = __hip_atomic_fetch_add(cnt, 1u, __ATOMIC_ACQ_REL,
                                              __HIP_MEMORY_SCOPE_AGENT);
    if (arrived == nblk - 1) {
      __hip_atomic_store(cnt, 0u, __ATOMIC_RELAXED, __HIP_MEMORY_SCOPE_AGENT);
      __hip_atomic_fetch_add(gen, 1u, __ATOMIC_ACQ_REL, __HIP_MEMORY_SCOPE_AGENT);
    } else {
      while (__hip_atomic_load(gen, __ATOMIC_ACQUIRE, __HIP_MEMORY_SCOPE_AGENT) == g)
        __builtin_amdgcn_s_sleep(2);
    }
  }
  __syncthreads();
}

// --------------------------- persistent recurrent kernel: W_hh pinned in LDS
// Block b owns hidden units [b*8, b*8+8): all four gate rows per unit, so the
// cell update is block-local. Only h (4KB) crosses blocks, via agent-scope L2.
__global__ void __launch_bounds__(256, 1)
lstm_recurrent(const float* __restrict__ Gih,
               const float* __restrict__ Whh,
               const float* __restrict__ b_ih,
               const float* __restrict__ b_hh,
               float* hb0, float* hb1, unsigned* sync,
               float* __restrict__ out) {
  extern __shared__ float smem[];
  float* Wsl = smem;                        // 32 x WSTRIDE  (W_hh slice)
  float* hsl = Wsl + 32 * WSTRIDE;          // H             (h staged per step)
  float* gsl = hsl + H;                     // 32 gate pre-activations
  float* csl = gsl + 32;                    // 8 cell states

  const int tid = threadIdx.x;
  const int j0  = blockIdx.x * UPB;

  // LDS row r = u*4+g  <->  W_hh row  g*H + (j0+u)   (g: 0=i,1=f,2=g,3=o)
  for (int idx = tid; idx < 32 * H; idx += blockDim.x) {
    int r = idx >> 10, k = idx & (H - 1);
    int u = r >> 2, g = r & 3;
    Wsl[r * WSTRIDE + k] = Whh[(long)(g * H + j0 + u) * H + k];
  }
  if (tid < UPB) csl[tid] = 0.f;
  __syncthreads();

  const int r    = tid >> 3;                // 32 rows x 8 threads each
  const int seg  = tid & 7;
  const int u_   = r >> 2, g_ = r & 3;
  const int grow = g_ * H + j0 + u_;
  const float bias = b_ih[grow] + b_hh[grow];
  const float* wrow = Wsl + r * WSTRIDE;

  float* hbuf[2] = { hb0, hb1 };

  for (int t = 0; t < S; ++t) {
    const float* hc = hbuf[t & 1];
    for (int i = tid; i < H; i += blockDim.x)   // device-scope: bypass stale WGP$
      hsl[i] = __hip_atomic_load(hc + i, __ATOMIC_RELAXED, __HIP_MEMORY_SCOPE_AGENT);
    __syncthreads();

    float acc = 0.f;
#pragma unroll 8
    for (int i = 0; i < H / 8; ++i) {
      int k = seg + (i << 3);
      acc = __builtin_fmaf(wrow[k], hsl[k], acc);
    }
    acc += __shfl_down(acc, 4, 8);
    acc += __shfl_down(acc, 2, 8);
    acc += __shfl_down(acc, 1, 8);
    if (seg == 0)
      gsl[r] = acc + Gih[(long)t * G4H + grow] + bias;
    __syncthreads();

    if (tid < UPB) {
      float gi = sigmoidf(gsl[tid * 4 + 0]);
      float gf = sigmoidf(gsl[tid * 4 + 1]);
      float gg = tanhf   (gsl[tid * 4 + 2]);
      float go = sigmoidf(gsl[tid * 4 + 3]);
      float c  = gf * csl[tid] + gi * gg;
      csl[tid] = c;
      float hn = go * tanhf(c);
      __hip_atomic_store(hbuf[(t + 1) & 1] + j0 + tid, hn,
                         __ATOMIC_RELAXED, __HIP_MEMORY_SCOPE_AGENT);
    }
    grid_sync(sync, sync + 1, NBLK_REC);
  }

  if (tid < UPB) {                            // S even -> final h in hbuf[0]
    float hfin = __hip_atomic_load(hbuf[S & 1] + j0 + tid,
                                   __ATOMIC_RELAXED, __HIP_MEMORY_SCOPE_AGENT);
    out[j0 + tid]         = hfin;             // out
    out[H + j0 + tid]     = hfin;             // h
    out[2 * H + j0 + tid] = csl[tid];         // c
  }
}

// ---------------------------------------------------------------- launch
extern "C" void kernel_launch(void* const* d_in, const int* in_sizes, int n_in,
                              void* d_out, int out_size, void* d_ws, size_t ws_size,
                              hipStream_t stream) {
  const int*   tok = (const int*)  d_in[0];
  const float* emb = (const float*)d_in[1];
  const float* Wih = (const float*)d_in[2];
  const float* Whh = (const float*)d_in[3];
  const float* bih = (const float*)d_in[4];
  const float* bhh = (const float*)d_in[5];
  float* out = (float*)d_out;

  char* ws = (char*)d_ws;
  float*          Gih = (float*)ws;                                       // 64 MB
  unsigned short* Xb  = (unsigned short*)(ws + (size_t)64 * 1024 * 1024); // 8 MB
  unsigned short* Wb  = (unsigned short*)(ws + (size_t)72 * 1024 * 1024); // 8 MB
  float*    hb0  = (float*)(ws + (size_t)80 * 1024 * 1024);
  float*    hb1  = hb0 + H;
  unsigned* sync = (unsigned*)(hb1 + H);

  init_state<<<(H + 255) / 256, 256, 0, stream>>>(hb0, hb1, sync);
  embed_relu_bf16<<<S, 256, 0, stream>>>(tok, emb, Xb);
  cvt_bf16<<<(G4H * E + 255) / 256, 256, 0, stream>>>(Wih, Wb, G4H * E);

  // 256 M-tiles x 64 N-strips = 16384 wave-tasks / 8 waves per block
  gemm_ih_wmma<<<(S / 16) * (G4H / 64) / 8, 256, 0, stream>>>(Xb, Wb, Gih);

  size_t lds = (size_t)(32 * WSTRIDE + H + 32 + UPB) * sizeof(float);     // ~136 KB
  lstm_recurrent<<<NBLK_REC, 256, lds, stream>>>(Gih, Whh, bih, bhh,
                                                 hb0, hb1, sync, out);
}
